// MemoryGraph_49460843381247
// MI455X (gfx1250) — compile-verified, matrix operand order checked by reading, not run on hardware
//
#include <hip/hip_runtime.h>
#include <math.h>

// Problem constants (must match the reference)
#define BS 8
#define GH 32
#define GW 32
#define NC (GH*GW)   // 1024
#define CN 64
#define DN 32
#define HS 128
#define HM 128

typedef __attribute__((ext_vector_type(16))) _Float16 v16h;
typedef __attribute__((ext_vector_type(8)))  float    v8f;

// Load a 16x32 f16 fragment (A-layout) from a row-major [rows][ld] LDS array.
// Per CDNA5 ISA (CDNA5 ISA, 16-bit A 16x32): lane 0-15 -> M=lane, K-halves
// 0..7 in VGPR0-3 and 16..23 in VGPR4-7; lanes 16-31 -> +8 on the K base.
// The B operand (K x 16) uses the symmetric layout with lane = column, so the
// same gather from a [N][K] (i.e. natural weight (out,in)) array produces B.
__device__ __forceinline__ v16h ldfrag(const _Float16* s, int row0, int ld,
                                       int kbase, int lane) {
  const _Float16* p = s + (size_t)(row0 + (lane & 15)) * ld + kbase
                        + ((lane & 16) ? 8 : 0);
  v16h f;
#pragma unroll
  for (int v = 0; v < 4; ++v) { f[2*v]   = p[2*v];      f[2*v+1]   = p[2*v+1]; }
#pragma unroll
  for (int v = 0; v < 4; ++v) { f[8+2*v] = p[16+2*v];   f[8+2*v+1] = p[16+2*v+1]; }
  return f;
}

__device__ __forceinline__ v8f wmma16(v16h a, v16h b, v8f c) {
  // D = A(16x32 f16) x B(32x16 f16) + C(16x16 f32)
  return __builtin_amdgcn_wmma_f32_16x16x32_f16(false, a, false, b,
                                                (short)0, c, false, false);
}

__device__ __forceinline__ float sigm(float x) { return 1.0f / (1.0f + expf(-x)); }

__global__ __launch_bounds__(128)
void memgraph_cell_kernel(
    const float* __restrict__ h,   const float* __restrict__ msg,
    const float* __restrict__ W,   const float* __restrict__ decay_logit,
    const float* __restrict__ bgl, const float* __restrict__ x,
    const float* __restrict__ nid, const int*   __restrict__ c2g,
    const float* __restrict__ w1r, const float* __restrict__ w1h,
    const float* __restrict__ b1,  const float* __restrict__ w2,
    const float* __restrict__ b2,  const float* __restrict__ mw1,
    const float* __restrict__ mb1, const float* __restrict__ mw2,
    const float* __restrict__ mb2, const float* __restrict__ injw,
    const float* __restrict__ injb,
    float* __restrict__ h_out,     float* __restrict__ msg_out)
{
  // LDS (~59 KB static). Buffers are phase-reused:
  //  sW:   W (A of GEMM1)            -> m_in = [h_new | neuron_id]
  //  sBm:  msg^T (B of GEMM1)        -> received (f16, [64][32])
  //  sAct: hidden [64][128]          -> mh [64][128]
  //  sWa:  w1r|w1h                   -> mw1
  //  sWb:  w2                        -> mw2
  __shared__ _Float16 sW  [CN * 64];
  __shared__ _Float16 sBm [DN * CN];
  __shared__ _Float16 sH  [CN * DN];
  __shared__ _Float16 sAct[CN * HS];
  __shared__ _Float16 sWa [HS * 64];
  __shared__ _Float16 sWb [DN * HS];
  __shared__ float sB1[HS], sB2[DN], sMB1[HM], sMB2[DN];
  __shared__ float sX[DN], sGate[4], sDecay[CN];

  const int tid  = threadIdx.x;
  const int lane = tid & 31;
  const int wv   = tid >> 5;
  const int bid  = blockIdx.x;
  const int b    = bid / NC;
  const int c    = bid - b * NC;
  const size_t cell  = (size_t)b * NC + c;
  const size_t baseA = cell * (size_t)(CN * DN);  // h / msg / outputs
  const size_t baseW = cell * (size_t)(CN * CN);

  // ---------------- stage ----------------
  for (int i = tid; i < CN * CN; i += 128) sW[i] = (_Float16)W[baseW + i];
  for (int i = tid; i < CN * DN; i += 128) {
    int j = i >> 5, d = i & 31;                  // msg row j, dim d
    sBm[d * CN + j] = (_Float16)msg[baseA + i];  // transpose: B wants [DN][CN]
    sH[i]           = (_Float16)h[baseA + i];
  }
  for (int i = tid; i < HS * DN; i += 128) {
    sWa[i]           = (_Float16)w1r[i];
    sWa[HS * DN + i] = (_Float16)w1h[i];
  }
  for (int i = tid; i < DN * HS; i += 128) sWb[i] = (_Float16)w2[i];
  if (tid < HS) { sB1[tid] = b1[tid]; sMB1[tid] = mb1[tid]; }
  if (tid < DN) { sB2[tid] = b2[tid]; sMB2[tid] = mb2[tid];
                  sX[tid]  = x[cell * DN + tid]; }
  if (tid < 4)  sGate[tid]  = sigm(bgl[cell * 4 + tid]);
  if (tid < CN) sDecay[tid] = sigm(decay_logit[cell * CN + tid]);
  __syncthreads();

  const int row0  = wv * 16;
  const int mBase = row0 + ((lane & 16) ? 8 : 0);
  const int d0    = lane & 15;

  // ---------------- phase 1: received = W @ msg ----------------
  v8f acc0 = {}, acc1 = {};
  {
    v16h a0  = ldfrag(sW,  row0, 64,  0, lane);
    v16h a1  = ldfrag(sW,  row0, 64, 32, lane);
    v16h b00 = ldfrag(sBm,    0, 64,  0, lane);
    v16h b01 = ldfrag(sBm,    0, 64, 32, lane);
    v16h b10 = ldfrag(sBm,   16, 64,  0, lane);
    v16h b11 = ldfrag(sBm,   16, 64, 32, lane);
    acc0 = wmma16(a0, b00, acc0);
    acc0 = wmma16(a1, b01, acc0);
    acc1 = wmma16(a0, b10, acc1);
    acc1 = wmma16(a1, b11, acc1);
  }

  // Corrections: inject (rows 0-3) and border (rows 8-11) live entirely in
  // wave 0's accumulator tile -> patch registers before writeback.
  if (wv == 0) {
    if (lane < 16) {                       // rows m = r (0..3): inject ports
      int g = c2g[c];
      const float* iwg = injw + (size_t)g * (4 * DN) * DN;
      const float* ibg = injb + (size_t)g * (4 * DN);
#pragma unroll
      for (int r = 0; r < 4; ++r) {
        float s0 = ibg[r * DN + d0];
        float s1 = ibg[r * DN + d0 + 16];
        const float* q0 = iwg + (size_t)(r * DN + d0)      * DN;
        const float* q1 = iwg + (size_t)(r * DN + d0 + 16) * DN;
#pragma unroll
        for (int i = 0; i < DN; ++i) { s0 += sX[i] * q0[i]; s1 += sX[i] * q1[i]; }
        acc0[r] += s0; acc1[r] += s1;
      }
    } else {                               // rows m = 8+r: border exchange
      int dd = lane - 16;
      int gy = c / GW, gx = c - (c / GW) * GW;
#pragma unroll
      for (int r = 0; r < 4; ++r) {
        int srcC = -1, srcRow = 0;
        if (r == 0 && gy > 0)      { srcC = c - GW; srcRow = 9;  }
        if (r == 1 && gy < GH - 1) { srcC = c + GW; srcRow = 8;  }
        if (r == 2 && gx > 0)      { srcC = c - 1;  srcRow = 11; }
        if (r == 3 && gx < GW - 1) { srcC = c + 1;  srcRow = 10; }
        if (srcC >= 0) {
          const float* mp = msg + ((size_t)b * NC + srcC) * (CN * DN) + srcRow * DN;
          float g = sGate[r];
          acc0[r] += g * mp[dd];
          acc1[r] += g * mp[dd + 16];
        }
      }
    }
  }
  __syncthreads();                          // everyone done reading msg^T

  // writeback received (f16) into the msg^T buffer, [64][32] row-major
  {
    _Float16* sRecv = sBm;
#pragma unroll
    for (int r = 0; r < 8; ++r) {
      sRecv[(mBase + r) * DN + d0]      = (_Float16)acc0[r];
      sRecv[(mBase + r) * DN + d0 + 16] = (_Float16)acc1[r];
    }
  }
  // neuron_id -> m_in columns 32..63 (W buffer is dead now)
  for (int i = tid; i < CN * DN; i += 128) {
    int m = i >> 5, d = i & 31;
    sW[m * 64 + 32 + d] = (_Float16)nid[(size_t)c * (CN * DN) + i];
  }
  __syncthreads();

  // ------- phase 2: hidden = tanh(recv@w1r^T + h@w1h^T + b1) -------
  {
    v16h aR = ldfrag(sBm, row0, DN, 0, lane);
    v16h aH = ldfrag(sH,  row0, DN, 0, lane);
#pragma unroll
    for (int n = 0; n < HS / 16; ++n) {
      v16h bR = ldfrag(sWa,           n * 16, DN, 0, lane);
      v16h bH = ldfrag(sWa + HS * DN, n * 16, DN, 0, lane);
      v8f acc = {};
      acc = wmma16(aH, bH, acc);
      acc = wmma16(aR, bR, acc);
      int col = n * 16 + d0;
      float bb = sB1[col];
#pragma unroll
      for (int r = 0; r < 8; ++r)
        sAct[(mBase + r) * HS + col] = (_Float16)tanhf(acc[r] + bb);
    }
  }

  // ------- phase 3: candidate = tanh(hidden@w2^T + b2); decay blend -------
  v8f cc0 = {}, cc1 = {};
#pragma unroll
  for (int k = 0; k < 4; ++k) {
    v16h aF = ldfrag(sAct, row0, HS, 32 * k, lane);
    v16h bA = ldfrag(sWb,     0, HS, 32 * k, lane);
    v16h bB = ldfrag(sWb,    16, HS, 32 * k, lane);
    cc0 = wmma16(aF, bA, cc0);
    cc1 = wmma16(aF, bB, cc1);
  }
#pragma unroll
  for (int r = 0; r < 8; ++r) {
    int m = mBase + r;
    float dec = sDecay[m];
    size_t gi = baseA + (size_t)m * DN;
    float c0 = tanhf(cc0[r] + sB2[d0]);
    float c1 = tanhf(cc1[r] + sB2[d0 + 16]);
    float hn0 = dec * h[gi + d0]      + (1.0f - dec) * c0;
    float hn1 = dec * h[gi + d0 + 16] + (1.0f - dec) * c1;
    h_out[gi + d0]      = hn0;
    h_out[gi + d0 + 16] = hn1;
    sW[m * 64 + d0]      = (_Float16)hn0;   // m_in columns 0..31
    sW[m * 64 + d0 + 16] = (_Float16)hn1;
  }
  __syncthreads();                          // all waves done with w1/w2
  for (int i = tid; i < HM * 64; i += 128) sWa[i] = (_Float16)mw1[i];
  for (int i = tid; i < DN * HM; i += 128) sWb[i] = (_Float16)mw2[i];
  __syncthreads();

  // ------- phase 4: mh = tanh(m_in @ mw1^T + mb1) -------
  {
    v16h a0 = ldfrag(sW, row0, 64,  0, lane);
    v16h a1 = ldfrag(sW, row0, 64, 32, lane);
#pragma unroll
    for (int n = 0; n < HM / 16; ++n) {
      v16h b0 = ldfrag(sWa, n * 16, 64,  0, lane);
      v16h b1f = ldfrag(sWa, n * 16, 64, 32, lane);
      v8f acc = {};
      acc = wmma16(a0, b0, acc);
      acc = wmma16(a1, b1f, acc);
      int col = n * 16 + d0;
      float bb = sMB1[col];
#pragma unroll
      for (int r = 0; r < 8; ++r)
        sAct[(mBase + r) * HM + col] = (_Float16)tanhf(acc[r] + bb);
    }
  }

  // ------- phase 5: msg_new = tanh(mh @ mw2^T + mb2) -------
  {
    v8f m0 = {}, m1 = {};
#pragma unroll
    for (int k = 0; k < 4; ++k) {
      v16h aF = ldfrag(sAct, row0, HM, 32 * k, lane);
      v16h b0 = ldfrag(sWb,     0, HM, 32 * k, lane);
      v16h b1f = ldfrag(sWb,    16, HM, 32 * k, lane);
      m0 = wmma16(aF, b0, m0);
      m1 = wmma16(aF, b1f, m1);
    }
#pragma unroll
    for (int r = 0; r < 8; ++r) {
      int m = mBase + r;
      size_t gi = baseA + (size_t)m * DN;
      msg_out[gi + d0]      = tanhf(m0[r] + sMB2[d0]);
      msg_out[gi + d0 + 16] = tanhf(m1[r] + sMB2[d0 + 16]);
    }
  }
}

extern "C" void kernel_launch(void* const* d_in, const int* in_sizes, int n_in,
                              void* d_out, int out_size, void* d_ws, size_t ws_size,
                              hipStream_t stream) {
  (void)in_sizes; (void)n_in; (void)d_ws; (void)ws_size; (void)out_size;
  const float* h    = (const float*)d_in[0];
  const float* msg  = (const float*)d_in[1];
  const float* W    = (const float*)d_in[2];
  const float* dec  = (const float*)d_in[3];
  const float* bgl  = (const float*)d_in[4];
  const float* x    = (const float*)d_in[5];
  const float* nid  = (const float*)d_in[6];
  const int*   c2g  = (const int*)d_in[7];
  const float* w1r  = (const float*)d_in[8];
  const float* w1h  = (const float*)d_in[9];
  const float* b1   = (const float*)d_in[10];
  const float* w2   = (const float*)d_in[11];
  const float* b2   = (const float*)d_in[12];
  const float* mw1  = (const float*)d_in[13];
  const float* mb1  = (const float*)d_in[14];
  const float* mw2  = (const float*)d_in[15];
  const float* mb2  = (const float*)d_in[16];
  const float* injw = (const float*)d_in[17];
  const float* injb = (const float*)d_in[18];

  float* h_out   = (float*)d_out;
  float* msg_out = h_out + (size_t)BS * NC * CN * DN;

  dim3 grid(BS * NC), block(128);
  hipLaunchKernelGGL(memgraph_cell_kernel, grid, block, 0, stream,
                     h, msg, W, dec, bgl, x, nid, c2g,
                     w1r, w1h, b1, w2, b2, mw1, mb1, mw2, mb2, injw, injb,
                     h_out, msg_out);
}
